// EventPairCompositionModel_69337952026904
// MI455X (gfx1250) — matrix operand order repaired and checked by reading, not compiled, not run on hardware
//
#include <hip/hip_runtime.h>
#include <hip/hip_bf16.h>

// ---------------------------------------------------------------------------
// EventPairCompositionModel forward, CDNA5 (gfx1250), wave32 + WMMA f16.
// B=512, N=128, C=4, E=300, V=50000, H1=512, H2=256, NF=8, K=11
// M = B + B*N = 66048 rows through the shared MLP.
//
// gemm1: one WG per 16-row tile; gather 16x1200 embeddings to LDS once;
//        each of 4 waves register-blocks 8 column tiles (acc[8]) so a single
//        A-fragment (2x ds_load_b128) feeds 8 v_wmma_f32_16x16x32_f16.
// gemm2: same scheme, 4 column tiles per wave, A straight from global f16.
// ---------------------------------------------------------------------------

typedef __attribute__((ext_vector_type(16))) _Float16 v16h;
typedef __attribute__((ext_vector_type(8)))  _Float16 v8h;
typedef __attribute__((ext_vector_type(8)))  float    v8f;

union AB { v16h v; v8h h[2]; };

#define BQ   512      // batch
#define NN   128      // context events
#define CC   4        // components
#define EE   300      // embed dim
#define K1   1200     // C*E
#define K1P  1216     // padded to multiple of 32 (38 chunks)
#define H1   512
#define H2   256
#define MTOT 66048    // BQ + BQ*NN
#define NFEAT 8
#define NKER 11

__constant__ float c_MUS[NKER]    = {1.0f, 0.9f, 0.7f, 0.5f, 0.3f, 0.1f, -0.1f, -0.3f, -0.5f, -0.7f, -0.9f};
__constant__ float c_SIGMAS[NKER] = {1e-3f, 0.1f, 0.1f, 0.1f, 0.1f, 0.1f, 0.1f, 0.1f, 0.1f, 0.1f, 0.1f};

// --------------------------- weight conversion -----------------------------
__global__ void convert_w1(const float* __restrict__ W1, _Float16* __restrict__ w1h) {
    int i = blockIdx.x * blockDim.x + threadIdx.x;
    if (i >= H1 * K1P) return;
    int n = i / K1P, k = i - n * K1P;
    w1h[i] = (k < K1) ? (_Float16)W1[n * K1 + k] : (_Float16)0.0f;
}

__global__ void convert_w2(const float* __restrict__ W2, _Float16* __restrict__ w2h) {
    int i = blockIdx.x * blockDim.x + threadIdx.x;
    if (i >= H2 * H1) return;
    w2h[i] = (_Float16)W2[i];
}

// ------------------------------- GEMM layer 1 ------------------------------
// Y1[M,512] = relu(X[M,1200] @ W1^T + b1), X gathered from event_table.
// Grid: (M/16); block 128 (4 waves); wave w covers columns [w*128, w*128+128).
__global__ __launch_bounds__(128)
void gemm1_kernel(const float* __restrict__ table,
                  const int*   __restrict__ batch_event,
                  const int*   __restrict__ batch_context,
                  const _Float16* __restrict__ w1h,
                  const float* __restrict__ b1,
                  _Float16* __restrict__ y1h)
{
    __shared__ __align__(16) _Float16 ldsA[16 * K1P];
    __shared__ int sidx[64];

    const int t       = threadIdx.x;
    const int rowBase = blockIdx.x * 16;
    const int w       = t >> 5;   // wave id 0..3

    // indices for the 16 rows of this tile (4 component ids per row)
    if (t < 64) {
        int r = t >> 2, c = t & 3;
        int gr = rowBase + r;
        sidx[t] = (gr < BQ) ? batch_event[gr * CC + c]
                            : batch_context[(gr - BQ) * CC + c];
    }
    __syncthreads();

    // cooperative gather: 16 rows x 1216 halves (zero-padded K tail), ONCE.
    for (int li = t; li < 16 * K1P; li += 128) {
        int r = li / K1P;
        int k = li - r * K1P;
        float v = 0.0f;
        if (k < K1) {
            int c = k / EE;
            int e = k - c * EE;
            v = table[(long)sidx[(r << 2) + c] * EE + e];
        }
        ldsA[li] = (_Float16)v;
    }
    __syncthreads();

    const int lane = t & 31;
    const int mn   = lane & 15;
    const int hi   = lane >> 4;

    v8f acc[8] = {};
    const _Float16* aptr  = ldsA + mn * K1P + hi * 8;
    const _Float16* bbase = w1h + (size_t)(w * 128 + mn) * K1P + hi * 8;

    for (int kc = 0; kc < K1P / 32; ++kc) {
        AB a;
        a.h[0] = *(const v8h*)(aptr + kc * 32);       // K 0-7   (+8 for hi lanes)
        a.h[1] = *(const v8h*)(aptr + kc * 32 + 16);  // K 16-23 (+8 for hi lanes)
        #pragma unroll
        for (int ct = 0; ct < 8; ++ct) {
            AB b;
            const _Float16* bp = bbase + (size_t)ct * 16 * K1P + kc * 32;
            b.h[0] = *(const v8h*)(bp);
            b.h[1] = *(const v8h*)(bp + 16);
            acc[ct] = __builtin_amdgcn_wmma_f32_16x16x32_f16(
                          false, a.v, false, b.v, (short)0, acc[ct], false, false);
        }
    }

    #pragma unroll
    for (int ct = 0; ct < 8; ++ct) {
        const int col  = w * 128 + ct * 16 + mn;
        const float bias = b1[col];
        #pragma unroll
        for (int j = 0; j < 8; ++j) {
            float v = acc[ct][j] + bias;
            v = v > 0.0f ? v : 0.0f;
            y1h[(size_t)(rowBase + j + hi * 8) * H1 + col] = (_Float16)v;
        }
    }
}

// ------------------------------- GEMM layer 2 ------------------------------
// Y2[M,256] = relu(Y1[M,512] @ W2^T + b2); A read straight from global f16.
// Grid: (M/16); block 128 (4 waves); wave w covers columns [w*64, w*64+64).
__global__ __launch_bounds__(128)
void gemm2_kernel(const _Float16* __restrict__ y1h,
                  const _Float16* __restrict__ w2h,
                  const float* __restrict__ b2,
                  float* __restrict__ y2)
{
    const int t       = threadIdx.x;
    const int rowBase = blockIdx.x * 16;
    const int w       = t >> 5;
    const int lane    = t & 31;
    const int mn      = lane & 15;
    const int hi      = lane >> 4;

    v8f acc[4] = {};
    const _Float16* aptr  = y1h + (size_t)(rowBase + mn) * H1 + hi * 8;
    const _Float16* bbase = w2h + (size_t)(w * 64 + mn) * H1 + hi * 8;

    for (int kc = 0; kc < H1 / 32; ++kc) {
        AB a;
        a.h[0] = *(const v8h*)(aptr + kc * 32);
        a.h[1] = *(const v8h*)(aptr + kc * 32 + 16);
        #pragma unroll
        for (int ct = 0; ct < 4; ++ct) {
            AB b;
            const _Float16* bp = bbase + (size_t)ct * 16 * H1 + kc * 32;
            b.h[0] = *(const v8h*)(bp);
            b.h[1] = *(const v8h*)(bp + 16);
            acc[ct] = __builtin_amdgcn_wmma_f32_16x16x32_f16(
                          false, a.v, false, b.v, (short)0, acc[ct], false, false);
        }
    }

    #pragma unroll
    for (int ct = 0; ct < 4; ++ct) {
        const int col  = w * 64 + ct * 16 + mn;
        const float bias = b2[col];
        #pragma unroll
        for (int j = 0; j < 8; ++j) {
            float v = acc[ct][j] + bias;
            y2[(size_t)(rowBase + j + hi * 8) * H2 + col] = v > 0.0f ? v : 0.0f;
        }
    }
}

// ---------------------- cosine + KNRM + head (one block/b) -----------------
__global__ __launch_bounds__(128)
void finalize_kernel(const float* __restrict__ y2,
                     const float* __restrict__ table,
                     const float* __restrict__ Wv,
                     const float* __restrict__ bv,
                     const float* __restrict__ Wc,
                     const float* __restrict__ bc,
                     const int*   __restrict__ batch_event,
                     const float* __restrict__ batch_features,
                     const float* __restrict__ batch_distances,
                     float* __restrict__ out)
{
    __shared__ float ev[H2];
    __shared__ float red[128];
    __shared__ float cosv[NN];
    __shared__ float kp[NKER];
    __shared__ float ext[17];
    __shared__ float inv_e_sh;

    const int b = blockIdx.x;
    const int t = threadIdx.x;

    // event representation + norm
    ev[t]       = y2[(size_t)b * H2 + t];
    ev[t + 128] = y2[(size_t)b * H2 + t + 128];
    __syncthreads();
    red[t] = ev[t] * ev[t] + ev[t + 128] * ev[t + 128];
    __syncthreads();
    for (int s = 64; s > 0; s >>= 1) {
        if (t < s) red[t] += red[t + s];
        __syncthreads();
    }
    if (t == 0) inv_e_sh = 1.0f / fmaxf(sqrtf(red[0]), 1e-12f);
    __syncthreads();
    const float inv_e = inv_e_sh;

    // cosine similarity vs context event t
    const float* crow = y2 + (size_t)(BQ + b * NN + t) * H2;
    float dot = 0.0f, cn = 0.0f;
    for (int i = 0; i < H2; ++i) {
        float c = crow[i];
        dot += c * ev[i];
        cn  += c * c;
    }
    cosv[t] = dot * inv_e * (1.0f / fmaxf(sqrtf(cn), 1e-12f));
    __syncthreads();

    // KNRM kernel pooling
    if (t < NKER) {
        float mu = c_MUS[t];
        float s2 = 2.0f * c_SIGMAS[t] * c_SIGMAS[t];
        float sum = 0.0f;
        for (int n = 0; n < NN; ++n) {
            float d = cosv[n] - mu;
            sum += expf(-(d * d) / s2);
        }
        kp[t] = logf(fmaxf(sum, 1e-10f)) * 0.01f;
    }

    // event-dependent Gaussian distance features (9 dims)
    if (t >= 32 && t < 32 + 9) {
        int j = t - 32;
        int pid = batch_event[b * CC + 1];
        const float* pr = table + (size_t)pid * EE;
        const float* wv = Wv + j * EE;
        float s = bv[j];
        for (int e = 0; e < EE; ++e) s += wv[e] * pr[e];
        float var = (s > 20.0f) ? s : log1pf(expf(s));   // softplus
        float d = batch_distances[b * 9 + j];
        ext[j] = expf(-(d * d) / var);
    }
    if (t >= 64 && t < 64 + NFEAT) {
        ext[9 + (t - 64)] = batch_features[b * NFEAT + (t - 64)];
    }
    __syncthreads();

    if (t == 0) {
        float score = bc[0];
        for (int i = 0; i < 17; ++i)   score += Wc[i] * ext[i];
        for (int k = 0; k < NKER; ++k) score += Wc[17 + k] * kp[k];
        out[b] = 1.0f / (1.0f + expf(-score));
    }
}

// ------------------------------- launcher ----------------------------------
extern "C" void kernel_launch(void* const* d_in, const int* in_sizes, int n_in,
                              void* d_out, int out_size, void* d_ws, size_t ws_size,
                              hipStream_t stream) {
    const float* table           = (const float*)d_in[0];
    const float* W1              = (const float*)d_in[1];
    const float* b1              = (const float*)d_in[2];
    const float* W2              = (const float*)d_in[3];
    const float* b2              = (const float*)d_in[4];
    const float* Wv              = (const float*)d_in[5];
    const float* bv              = (const float*)d_in[6];
    const float* Wc              = (const float*)d_in[7];
    const float* bc              = (const float*)d_in[8];
    const int*   batch_event     = (const int*)d_in[9];
    const float* batch_features  = (const float*)d_in[10];
    const float* batch_distances = (const float*)d_in[11];
    const int*   batch_context   = (const int*)d_in[12];
    // d_in[13] = batch_slots: unused in the forward pass.

    // workspace layout (all offsets 256B aligned)
    char* ws = (char*)d_ws;
    const size_t o_w1h = 0;                                       // 512*1216*2  = 1,245,184
    const size_t o_w2h = o_w1h + (size_t)H1 * K1P * 2;            // + 262,144
    const size_t o_y1h = o_w2h + (size_t)H2 * H1 * 2;             // = 1,507,328
    const size_t o_y2  = o_y1h + (size_t)MTOT * H1 * 2;           // + 67,633,152
    _Float16* w1h = (_Float16*)(ws + o_w1h);
    _Float16* w2h = (_Float16*)(ws + o_w2h);
    _Float16* y1h = (_Float16*)(ws + o_y1h);
    float*    y2  = (float*)(ws + o_y2);                          // + 67,633,152 -> ~137 MB total

    convert_w1<<<(H1 * K1P + 255) / 256, 256, 0, stream>>>(W1, w1h);
    convert_w2<<<(H2 * H1 + 255) / 256, 256, 0, stream>>>(W2, w2h);

    gemm1_kernel<<<dim3(MTOT / 16), 128, 0, stream>>>(
        table, batch_event, batch_context, w1h, b1, y1h);

    gemm2_kernel<<<dim3(MTOT / 16), 128, 0, stream>>>(
        y1h, w2h, b2, y2);

    finalize_kernel<<<BQ, 128, 0, stream>>>(
        y2, table, Wv, bv, Wc, bc, batch_event, batch_features,
        batch_distances, (float*)d_out);
}
